// MultiHeadAttentionBlock_41137196761649
// MI455X (gfx1250) — compile-verified, matrix operand order checked by reading, not run on hardware
//
#include <hip/hip_runtime.h>

// MultiHeadAttentionBlock for MI455X (gfx1250, wave32, WMMA).
// Shapes: x[8,1024,512], Wq/Wk/Wv/Wo[512,512], lengths[8].
// d_out = out[8,1024,512] fp32 ++ alpha[8,8,1024,1024] fp32.
// Workspace (needs 24 MB): q,k as f16 [b][h][s][64]; v stored transposed
// [b][h][d][s]; ctx reuses q region.

#define B_  8
#define S_  1024
#define H_  8
#define DH_ 64
#define DM_ 512

typedef __attribute__((ext_vector_type(16))) _Float16 v16h;
typedef __attribute__((ext_vector_type(8)))  _Float16 h8;
typedef __attribute__((ext_vector_type(8)))  float    v8f;
typedef __attribute__((ext_vector_type(4)))  float    f4;

static __device__ __forceinline__ v8f wmma16x16x32(v16h a, v16h b, v8f c) {
  // 8 args: (neg_a, A, neg_b, B, c_mod, C, reuse_a, reuse_b)
  return __builtin_amdgcn_wmma_f32_16x16x32_f16(false, a, false, b, (short)0, c,
                                                false, false);
}

static __device__ __forceinline__ v16h cat_h8(h8 lo, h8 hi) {
  return __builtin_shufflevector(lo, hi, 0, 1, 2, 3, 4, 5, 6, 7, 8, 9, 10, 11,
                                 12, 13, 14, 15);
}

// A-fragment (16x32 f16, MxK) from row-major storage (ld elems between rows).
// Per ISA 7.12.2: lanes 0-15 hold row=lane, K = {0..7, 16..23}+k0 ;
//                 lanes 16-31 hold row=lane-16, K = {8..15, 24..31}+k0.
// Also serves as the B-operand load when B is available transposed
// (B(k,n) = src[n*ld + k]); "row" then plays the role of the N column.
// Both chunks are contiguous 16B -> b128 loads. Caller guarantees alignment.
static __device__ __forceinline__ v16h ldfrag_rowmajor_f16(
    const _Float16* __restrict__ src, int ld, int k0) {
  const int lane = threadIdx.x & 31;
  const int row  = lane & 15;
  const int koff = (lane & 16) ? 8 : 0;
  const _Float16* p = src + (size_t)row * ld + (k0 + koff);
  const h8 lo = *(const h8*)p;
  const h8 hi = *(const h8*)(p + 16);
  return cat_h8(lo, hi);
}

// Same fragment, converting from fp32 source (used for alpha @ V).
static __device__ __forceinline__ v16h ldfrag_rowmajor_f32(
    const float* __restrict__ src, int ld, int k0) {
  const int lane = threadIdx.x & 31;
  const int row  = lane & 15;
  const int koff = (lane & 16) ? 8 : 0;
  const float* p = src + (size_t)row * ld + (k0 + koff);
  const v8f lo = *(const v8f*)p;          // 32B-aligned by construction
  const v8f hi = *(const v8f*)(p + 16);
  return cat_h8(__builtin_convertvector(lo, h8),
                __builtin_convertvector(hi, h8));
}

// ---------------------------------------------------------------------------
// Kernel 1: Y(f16) = X[8192x512](f32) @ W[512x512](f32), head-major output.
// TRANSPOSE_OUT=false: Y[((b*H+h)<<16) + (s<<6) + d]   (q, k)
// TRANSPOSE_OUT=true : Y[((b*H+h)<<16) + (d<<10) + s]  (v transposed)
// Grid 1024 (128 m-blocks x 8 n-blocks), 256 threads (8 waves), 64x64 tile.
// W tile staged TRANSPOSED in LDS so B-fragments are contiguous b128 loads;
// the k-strided W fill is lane-coalesced and L2-resident (W is 1 MB).
// ---------------------------------------------------------------------------
template <bool TRANSPOSE_OUT>
__global__ __launch_bounds__(256) void k_proj(const float* __restrict__ X,
                                              const float* __restrict__ W,
                                              _Float16* __restrict__ Y) {
  __shared__ __align__(16) _Float16 As[64][40];   // [m][k], stride 80B
  __shared__ __align__(16) _Float16 BsT[64][40];  // [n][k], stride 80B
  const int m0   = (blockIdx.x >> 3) * 64;
  const int n0   = (blockIdx.x & 7) * 64;
  const int tid  = threadIdx.x;
  const int wave = tid >> 5;
  const int mt   = wave >> 1;       // 0..3
  const int ntb  = (wave & 1) * 2;  // 0 or 2
  const int ar = tid >> 2, ac = (tid & 3) * 8;   // A fill: 8 f32 along k
  const int bn = tid & 63, bk = (tid >> 6) * 8;  // B fill: 8 k's, lane-coal n
  v8f acc0 = {}, acc1 = {};
  for (int k0 = 0; k0 < DM_; k0 += 32) {
    const v8f av = *(const v8f*)&X[(size_t)(m0 + ar) * DM_ + k0 + ac];
    *(h8*)&As[ar][ac] = __builtin_convertvector(av, h8);
    h8 wv;
#pragma unroll
    for (int i = 0; i < 8; ++i)
      wv[i] = (_Float16)W[(size_t)(k0 + bk + i) * DM_ + n0 + bn];
    *(h8*)&BsT[bn][bk] = wv;
    __syncthreads();
    if (k0 + 32 < DM_)  // hint next A tile into cache (global_prefetch_b8)
      __builtin_prefetch(&X[(size_t)(m0 + ar) * DM_ + k0 + 32], 0, 0);
    const v16h a  = ldfrag_rowmajor_f16(&As[mt * 16][0], 40, 0);
    const v16h b0 = ldfrag_rowmajor_f16(&BsT[ntb * 16][0], 40, 0);
    const v16h b1 = ldfrag_rowmajor_f16(&BsT[ntb * 16 + 16][0], 40, 0);
    acc0 = wmma16x16x32(a, b0, acc0);
    acc1 = wmma16x16x32(a, b1, acc1);
    __syncthreads();
  }
  const int lane = tid & 31;
  const int col  = lane & 15;
  const int r0   = (lane & 16) ? 8 : 0;
#pragma unroll
  for (int j = 0; j < 2; ++j) {
    const v8f acc = j ? acc1 : acc0;
    const int gn  = n0 + (ntb + j) * 16 + col;
    const int h = gn >> 6, d = gn & 63;
#pragma unroll
    for (int r = 0; r < 8; ++r) {
      const int gm = m0 + mt * 16 + r0 + r;
      const int b = gm >> 10, s = gm & 1023;
      const size_t base = (size_t)(b * H_ + h) << 16;
      if (TRANSPOSE_OUT)
        Y[base + ((size_t)d << 10) + s] = (_Float16)acc[r];
      else
        Y[base + ((size_t)s << 6) + d] = (_Float16)acc[r];
    }
  }
}

// ---------------------------------------------------------------------------
// Kernel 2: e = scale * Q K^T, masked -> alpha region (fp32).
// Grid 16384 = 64 (b,h) x 16 row-blocks x 16 col-blocks; 128 threads (4 waves).
// Each wave: one 16-row tile x four 16-col tiles; K=64 as two K=32 WMMAs.
// B = K^T so B(k,n) = K[n][k]: row-major fragment pattern on K rows (b128).
// ---------------------------------------------------------------------------
__global__ __launch_bounds__(128) void k_scores(const _Float16* __restrict__ qh,
                                                const _Float16* __restrict__ kh,
                                                const int* __restrict__ lengths,
                                                float* __restrict__ alpha) {
  const int cb = blockIdx.x & 15;
  const int rb = (blockIdx.x >> 4) & 15;
  const int bh = blockIdx.x >> 8;  // 0..63
  const int b  = bh >> 3;
  const int len = lengths[b];
  const _Float16* Q = qh + ((size_t)bh << 16);  // [1024][64]
  const _Float16* K = kh + ((size_t)bh << 16);
  float* A = alpha + ((size_t)bh << 20);        // [1024][1024]
  const int wave = threadIdx.x >> 5;
  const int m0   = rb * 64 + wave * 16;
  const v16h a0 = ldfrag_rowmajor_f16(Q + (size_t)m0 * DH_, DH_, 0);
  const v16h a1 = ldfrag_rowmajor_f16(Q + (size_t)m0 * DH_, DH_, 32);
  const int lane = threadIdx.x & 31;
  const int col  = lane & 15;
  const int r0   = (lane & 16) ? 8 : 0;
  for (int nt = 0; nt < 4; ++nt) {
    const int n0 = cb * 64 + nt * 16;
    const v16h b0 = ldfrag_rowmajor_f16(K + (size_t)n0 * DH_, DH_, 0);
    const v16h b1 = ldfrag_rowmajor_f16(K + (size_t)n0 * DH_, DH_, 32);
    v8f d = {};
    d = wmma16x16x32(a0, b0, d);
    d = wmma16x16x32(a1, b1, d);
    const int  gn   = n0 + col;
    const bool cval = (gn <= len);  // inclusive, as in reference
#pragma unroll
    for (int r = 0; r < 8; ++r) {
      const int gm = m0 + r0 + r;
      const float e = (cval && (gm <= len)) ? d[r] * 0.125f : -1e10f;
      A[((size_t)gm << 10) + gn] = e;
    }
  }
}

// ---------------------------------------------------------------------------
// Kernel 3: row softmax in place over the 1024-wide alpha rows.
// One wave32 per row; 8 waves per block; grid 8192 (65536 rows total).
// float4 accesses: 8x b128 per lane, fully coalesced.
// ---------------------------------------------------------------------------
__global__ __launch_bounds__(256) void k_softmax(float* __restrict__ alpha) {
  const int row  = blockIdx.x * 8 + (threadIdx.x >> 5);
  const int lane = threadIdx.x & 31;
  float* p = alpha + ((size_t)row << 10);
  f4 v[8];
  float m = -3.0e38f;
#pragma unroll
  for (int i = 0; i < 8; ++i) {
    v[i] = *(const f4*)(p + (size_t)i * 128 + lane * 4);
#pragma unroll
    for (int j = 0; j < 4; ++j) m = fmaxf(m, v[i][j]);
  }
#pragma unroll
  for (int off = 16; off > 0; off >>= 1) m = fmaxf(m, __shfl_xor(m, off, 32));
  float s = 0.f;
#pragma unroll
  for (int i = 0; i < 8; ++i) {
#pragma unroll
    for (int j = 0; j < 4; ++j) {
      v[i][j] = __expf(v[i][j] - m);
      s += v[i][j];
    }
  }
#pragma unroll
  for (int off = 16; off > 0; off >>= 1) s += __shfl_xor(s, off, 32);
  const float inv = 1.0f / s;
#pragma unroll
  for (int i = 0; i < 8; ++i) {
#pragma unroll
    for (int j = 0; j < 4; ++j) v[i][j] *= inv;
    *(f4*)(p + (size_t)i * 128 + lane * 4) = v[i];
  }
}

// ---------------------------------------------------------------------------
// Kernel 4: ctx[8192x512](f16) = alpha[1024x1024](f32) @ V[1024x64](f16).
// V is stored transposed (vt[b][h][d][s]) so B(k,n) = VT[n][k]: contiguous
// row-major fragment loads. Grid 1024 = 64 (b,h) x 16 row-blocks; 128 thr.
// ---------------------------------------------------------------------------
__global__ __launch_bounds__(128) void k_ctx(const float* __restrict__ alpha,
                                             const _Float16* __restrict__ vt,
                                             _Float16* __restrict__ ctx) {
  const int rb = blockIdx.x & 15;
  const int bh = blockIdx.x >> 4;
  const int b = bh >> 3, h = bh & 7;
  const float* A     = alpha + ((size_t)bh << 20);
  const _Float16* VT = vt + ((size_t)bh << 16);  // [64][1024]
  const int wave = threadIdx.x >> 5;
  const int m0   = rb * 64 + wave * 16;
  v8f acc[4] = {{}, {}, {}, {}};
  for (int k0 = 0; k0 < S_; k0 += 32) {
    const v16h a = ldfrag_rowmajor_f32(A + (size_t)m0 * S_, S_, k0);
#pragma unroll
    for (int nt = 0; nt < 4; ++nt) {
      const v16h bf = ldfrag_rowmajor_f16(VT + ((size_t)(nt * 16) << 10), S_, k0);
      acc[nt] = wmma16x16x32(a, bf, acc[nt]);
    }
  }
  const int lane = threadIdx.x & 31;
  const int col  = lane & 15;
  const int r0   = (lane & 16) ? 8 : 0;
#pragma unroll
  for (int nt = 0; nt < 4; ++nt) {
#pragma unroll
    for (int r = 0; r < 8; ++r) {
      const int s = m0 + r0 + r;
      ctx[(size_t)(b * S_ + s) * DM_ + h * DH_ + nt * 16 + col] =
          (_Float16)acc[nt][r];
    }
  }
}

// ---------------------------------------------------------------------------
// Kernel 5: out(fp32) = ctx[8192x512](f16) @ Wo[512x512](f32).
// Same tiling as k_proj (transposed B staging); row-major fp32 store.
// ---------------------------------------------------------------------------
__global__ __launch_bounds__(256) void k_outproj(const _Float16* __restrict__ C,
                                                 const float* __restrict__ Wo,
                                                 float* __restrict__ out) {
  __shared__ __align__(16) _Float16 As[64][40];
  __shared__ __align__(16) _Float16 BsT[64][40];
  const int m0   = (blockIdx.x >> 3) * 64;
  const int n0   = (blockIdx.x & 7) * 64;
  const int tid  = threadIdx.x;
  const int wave = tid >> 5;
  const int mt   = wave >> 1;
  const int ntb  = (wave & 1) * 2;
  const int ar = tid >> 2, ac = (tid & 3) * 8;
  const int bn = tid & 63, bk = (tid >> 6) * 8;
  v8f acc0 = {}, acc1 = {};
  for (int k0 = 0; k0 < DM_; k0 += 32) {
    *(h8*)&As[ar][ac] = *(const h8*)&C[(size_t)(m0 + ar) * DM_ + k0 + ac];
    h8 wv;
#pragma unroll
    for (int i = 0; i < 8; ++i)
      wv[i] = (_Float16)Wo[(size_t)(k0 + bk + i) * DM_ + n0 + bn];
    *(h8*)&BsT[bn][bk] = wv;
    __syncthreads();
    const v16h a  = ldfrag_rowmajor_f16(&As[mt * 16][0], 40, 0);
    const v16h b0 = ldfrag_rowmajor_f16(&BsT[ntb * 16][0], 40, 0);
    const v16h b1 = ldfrag_rowmajor_f16(&BsT[ntb * 16 + 16][0], 40, 0);
    acc0 = wmma16x16x32(a, b0, acc0);
    acc1 = wmma16x16x32(a, b1, acc1);
    __syncthreads();
  }
  const int lane = tid & 31;
  const int col  = lane & 15;
  const int r0   = (lane & 16) ? 8 : 0;
#pragma unroll
  for (int j = 0; j < 2; ++j) {
    const v8f acc = j ? acc1 : acc0;
    const int gn  = n0 + (ntb + j) * 16 + col;
#pragma unroll
    for (int r = 0; r < 8; ++r) {
      const int gm = m0 + mt * 16 + r0 + r;
      out[(size_t)gm * DM_ + gn] = acc[r];
    }
  }
}

// ---------------------------------------------------------------------------
extern "C" void kernel_launch(void* const* d_in, const int* in_sizes, int n_in,
                              void* d_out, int out_size, void* d_ws,
                              size_t ws_size, hipStream_t stream) {
  (void)in_sizes; (void)n_in; (void)out_size; (void)ws_size;
  const float* x       = (const float*)d_in[0];
  const int*   lengths = (const int*)d_in[1];
  const float* Wq      = (const float*)d_in[2];
  const float* Wk      = (const float*)d_in[3];
  const float* Wv      = (const float*)d_in[4];
  const float* Wo      = (const float*)d_in[5];

  float* out   = (float*)d_out;                // [8192][512]
  float* alpha = out + (size_t)B_ * S_ * DM_;  // [64][1024][1024]

  const size_t QKV = (size_t)B_ * H_ * S_ * DH_;  // 4M f16 elems each
  _Float16* qh  = (_Float16*)d_ws;                // [b][h][s][64]
  _Float16* kh  = qh + QKV;                       // [b][h][s][64]
  _Float16* vt  = kh + QKV;                       // [b][h][d][s] (transposed)
  _Float16* ctx = qh;  // q is dead after k_scores; reuse as [8192][512]

  k_proj<false><<<dim3(1024), dim3(256), 0, stream>>>(x, Wq, qh);
  k_proj<false><<<dim3(1024), dim3(256), 0, stream>>>(x, Wk, kh);
  k_proj<true><<<dim3(1024), dim3(256), 0, stream>>>(x, Wv, vt);
  k_scores<<<dim3(16384), dim3(128), 0, stream>>>(qh, kh, lengths, alpha);
  k_softmax<<<dim3(8192), dim3(256), 0, stream>>>(alpha);
  k_ctx<<<dim3(1024), dim3(128), 0, stream>>>(alpha, vt, ctx);
  k_outproj<<<dim3(1024), dim3(256), 0, stream>>>(ctx, Wo, out);
}
